// GCN_link_prediction_63565515981283
// MI455X (gfx1250) — compile-verified, matrix operand order checked by reading, not run on hardware
//
#include <hip/hip_runtime.h>
#include <hip/hip_bf16.h>
#include <stdint.h>

// ---------------------------------------------------------------------------
// GCN link prediction for MI455X (gfx1250, wave32, WMMA + TDM).
// Pipeline:
//   deg -> dinv -> cvt(x,W1^T,W2^T to bf16)
//   -> WMMA GEMM1 (TDM-staged B in LDS, bf16 -> f32 acc -> bf16)
//   -> agg1 init (self loop) -> agg1 edge scatter (f32 atomics, L2-resident)
//   -> bias+relu+cvt -> WMMA GEMM2 (TDM-staged B) -> agg2 init/scatter
//   -> decode (edge dot products)
// ---------------------------------------------------------------------------

#define N_NODES 50000
#define N_EDGES 1600000
#define N_LABEL 200000
#define IN_C    128
#define HID1    256
#define HID2    32

typedef __attribute__((ext_vector_type(16))) __bf16 v16bf;
typedef __attribute__((ext_vector_type(8)))  __bf16 v8bf;
typedef __attribute__((ext_vector_type(8)))  float  v8f;
typedef unsigned int u32x4 __attribute__((ext_vector_type(4)));
typedef int          i32x8 __attribute__((ext_vector_type(8)));
typedef int          i32x4 __attribute__((ext_vector_type(4)));

// ----------------------------- degree / dinv -------------------------------

__global__ void k_deg_init(float* __restrict__ deg) {
    int i = blockIdx.x * blockDim.x + threadIdx.x;
    if (i < N_NODES) deg[i] = 1.0f;            // self loop
}

__global__ void k_deg_edges(const int* __restrict__ ei, float* __restrict__ deg) {
    int e = blockIdx.x * blockDim.x + threadIdx.x;
    if (e < N_EDGES) {
        int d = ei[N_EDGES + e];               // dst
        atomicAdd(&deg[d], 1.0f);
    }
}

__global__ void k_dinv(const float* __restrict__ deg, float* __restrict__ dinv) {
    int i = blockIdx.x * blockDim.x + threadIdx.x;
    if (i < N_NODES) dinv[i] = rsqrtf(deg[i]); // deg >= 1 always
}

// ----------------------------- conversions ---------------------------------

__global__ void k_cvt_bf16(const float* __restrict__ in, __bf16* __restrict__ out, int n) {
    int i = blockIdx.x * blockDim.x + threadIdx.x;
    if (i < n) out[i] = (__bf16)in[i];
}

// W [K,N] row-major f32  ->  WT [N,K] row-major bf16 (column of W contiguous)
__global__ void k_transpose_cvt(const float* __restrict__ W, __bf16* __restrict__ WT,
                                int K, int N) {
    int i = blockIdx.x * blockDim.x + threadIdx.x;
    if (i < K * N) {
        int k = i / N, n = i - k * N;
        WT[n * K + k] = (__bf16)W[i];
    }
}

// ----------------------------- WMMA GEMM -----------------------------------
// C[M,N](bf16) = A[M,K](bf16) * B[K,N], with B supplied as BT[N,K] (bf16).
// The whole BT (N*K bf16; 64KB for layer 1, 16KB for layer 2) is DMA'd into
// LDS once per block by the Tensor Data Mover, then all 8 waves of the block
// feed their B fragments from LDS (ds_load_b128) while A streams from global.
// One wave computes one 16x16 tile; EXEC is all-1s inside every WMMA.
template <int K, int N>
__global__ void k_wmma_gemm(const __bf16* __restrict__ A,
                            const __bf16* __restrict__ BT,
                            __bf16* __restrict__ C,
                            int tilesM, int totalTiles) {
    __shared__ __bf16 Bs[N * K];

    // --- TDM: stage BT[N][K] (row-major, bf16) into LDS, one issue per block.
    if (threadIdx.x < 32) {                    // wave 0 only (EXEC ignored by TDM)
        uint64_t ga = (uint64_t)(uintptr_t)BT;
        u32x4 g0;
        g0.x = 1u;                             // count=1, is_restore=0, gather off
        g0.y = 0u;                             // lds_addr (Bs is the only LDS alloc)
        g0.z = (uint32_t)ga;                   // global_addr[31:0]  (bits 95:64)
        g0.w = (uint32_t)((ga >> 32) & 0x01FFFFFFu) | (2u << 30);  // addr[56:32] | type=2
        i32x8 g1;
        g1[0] = 0x00010000;                    // workgroup_mask=0, data_size=1 (2B)
        g1[1] = (K & 0xFFFF) << 16;            // tensor_dim0[15:0]
        g1[2] = (K >> 16) | ((N & 0xFFFF) << 16);   // tensor_dim0[31:16] | tensor_dim1[15:0]
        g1[3] = (N >> 16) | ((K & 0xFFFF) << 16);   // tensor_dim1[31:16] | tile_dim0
        g1[4] = N & 0xFFFF;                    // tile_dim1 (tile_dim2 = 0 -> 2D)
        g1[5] = K;                             // tensor_dim0_stride[31:0]
        g1[6] = 0;                             // stride0[47:32] | stride1[15:0]
        g1[7] = 0;                             // stride1[47:16]
        i32x4 g2 = {0, 0, 0, 0};
        i32x4 g3 = {0, 0, 0, 0};
        i32x8 g4 = {0, 0, 0, 0, 0, 0, 0, 0};   // 6-arg toolchain: extra group
        __builtin_amdgcn_tensor_load_to_lds(g0, g1, g2, g3, g4, 0);
        __builtin_amdgcn_s_wait_tensorcnt(0);
    }
    __syncthreads();

    int wave = (blockIdx.x * blockDim.x + threadIdx.x) >> 5;
    if (wave < totalTiles) {                   // wave-uniform guard
        int lane  = threadIdx.x & 31;
        int tM    = wave % tilesM;
        int tN    = wave / tilesM;
        int half  = lane >> 4;                 // 0: K blocks {0..7,16..23}, 1: {8..15,24..31}
        int koff  = half << 3;
        int arow  = tM * 16 + (lane & 15);     // A row per lane
        int bcol  = tN * 16 + (lane & 15);     // B column per lane (mirrored layout)

        v8f acc = {};
#pragma unroll
        for (int kk = 0; kk < K; kk += 32) {
            const __bf16* ap = A  + (size_t)arow * K + kk + koff;
            const __bf16* bp = Bs + (size_t)bcol * K + kk + koff;   // LDS
            v8bf a0 = *(const v8bf*)ap;
            v8bf a1 = *(const v8bf*)(ap + 16);
            v8bf b0 = *(const v8bf*)bp;
            v8bf b1 = *(const v8bf*)(bp + 16);
            v16bf av = __builtin_shufflevector(a0, a1, 0,1,2,3,4,5,6,7,8,9,10,11,12,13,14,15);
            v16bf bv = __builtin_shufflevector(b0, b1, 0,1,2,3,4,5,6,7,8,9,10,11,12,13,14,15);
            acc = __builtin_amdgcn_wmma_f32_16x16x32_bf16(
                      /*neg_a=*/false, av, /*neg_b=*/false, bv,
                      /*c_mod=*/(short)0, acc, /*reuse_a=*/false, /*reuse_b=*/false);
        }

        // C/D layout: VGPR r -> row r (lanes 0-15) / row r+8 (lanes 16-31)
        int rbase = tM * 16 + (half << 3);
        int col   = tN * 16 + (lane & 15);
#pragma unroll
        for (int r = 0; r < 8; ++r)
            C[(size_t)(rbase + r) * N + col] = (__bf16)acc[r];
    }
}

// ------------------------- layer-1 aggregation -----------------------------

// agg[n][c] = dinv[n]^2 * h[n][c]   (self-loop term; also initializes buffer)
__global__ void k_agg1_init(const __bf16* __restrict__ hbf,
                            const float* __restrict__ dinv,
                            float* __restrict__ agg) {
    int i = blockIdx.x * blockDim.x + threadIdx.x;
    if (i < N_NODES * HID1) {
        int n = i >> 8;                        // / 256
        float di = dinv[n];
        agg[i] = (float)hbf[i] * di * di;
    }
}

// one wave per edge; lane l handles channels [8l, 8l+8) via a 16B bf16 gather
__global__ void k_agg1_edges(const int* __restrict__ ei,
                             const float* __restrict__ dinv,
                             const __bf16* __restrict__ hbf,
                             float* __restrict__ agg) {
    int wave = (blockIdx.x * blockDim.x + threadIdx.x) >> 5;
    int lane = threadIdx.x & 31;
    int s = ei[wave];
    int d = ei[N_EDGES + wave];
    float coef = dinv[s] * dinv[d];
    const v8bf hv = *(const v8bf*)(hbf + (size_t)s * HID1 + lane * 8);
    float* dp = agg + (size_t)d * HID1 + lane * 8;
#pragma unroll
    for (int j = 0; j < 8; ++j)
        atomicAdd(dp + j, (float)hv[j] * coef);
}

// h1 = relu(agg + b1), stored bf16 for GEMM2
__global__ void k_post1(const float* __restrict__ agg,
                        const float* __restrict__ b1,
                        __bf16* __restrict__ h1bf) {
    int i = blockIdx.x * blockDim.x + threadIdx.x;
    if (i < N_NODES * HID1) {
        int c = i & (HID1 - 1);
        float v = agg[i] + b1[c];
        h1bf[i] = (__bf16)fmaxf(v, 0.0f);
    }
}

// ------------------------- layer-2 aggregation -----------------------------

// z[n][c] = dinv[n]^2 * zpre[n][c] + b2[c]
__global__ void k_agg2_init(const __bf16* __restrict__ zprebf,
                            const float* __restrict__ dinv,
                            const float* __restrict__ b2,
                            float* __restrict__ z) {
    int i = blockIdx.x * blockDim.x + threadIdx.x;
    if (i < N_NODES * HID2) {
        int n = i >> 5;                        // / 32
        int c = i & (HID2 - 1);
        float di = dinv[n];
        z[i] = (float)zprebf[i] * di * di + b2[c];
    }
}

// one wave per edge; lane == channel (HID2 == 32 == wave width)
__global__ void k_agg2_edges(const int* __restrict__ ei,
                             const float* __restrict__ dinv,
                             const __bf16* __restrict__ zprebf,
                             float* __restrict__ z) {
    int wave = (blockIdx.x * blockDim.x + threadIdx.x) >> 5;
    int lane = threadIdx.x & 31;
    int s = ei[wave];
    int d = ei[N_EDGES + wave];
    float coef = dinv[s] * dinv[d];
    float m = (float)zprebf[(size_t)s * HID2 + lane] * coef;
    atomicAdd(z + (size_t)d * HID2 + lane, m);
}

// ------------------------------ decode -------------------------------------

__global__ void k_decode(const int* __restrict__ eli,
                         const float* __restrict__ z,
                         float* __restrict__ out) {
    int t = blockIdx.x * blockDim.x + threadIdx.x;
    if (t < N_LABEL) {
        int a = eli[t];
        int b = eli[N_LABEL + t];
        const float4* za = (const float4*)(z + (size_t)a * HID2);
        const float4* zb = (const float4*)(z + (size_t)b * HID2);
        __builtin_prefetch(zb, 0, 0);          // global_prefetch_b8
        float acc = 0.0f;
#pragma unroll
        for (int i = 0; i < HID2 / 4; ++i) {
            float4 va = za[i], vb = zb[i];
            acc += va.x * vb.x + va.y * vb.y + va.z * vb.z + va.w * vb.w;
        }
        out[t] = acc;
    }
}

// ------------------------------ launcher -----------------------------------

extern "C" void kernel_launch(void* const* d_in, const int* in_sizes, int n_in,
                              void* d_out, int out_size, void* d_ws, size_t ws_size,
                              hipStream_t stream) {
    const float* x   = (const float*)d_in[0];   // [50000,128]
    const int*   ei  = (const int*)  d_in[1];   // [2,1600000]
    const int*   eli = (const int*)  d_in[2];   // [2,200000]
    const float* W1  = (const float*)d_in[3];   // [128,256]
    const float* b1  = (const float*)d_in[4];   // [256]
    const float* W2  = (const float*)d_in[5];   // [256,32]
    const float* b2  = (const float*)d_in[6];   // [32]
    float* out = (float*)d_out;                 // [200000]

    // workspace carve-up (256B aligned)
    char* ws = (char*)d_ws;
    size_t off = 0;
    auto carve = [&](size_t bytes) {
        size_t o = off;
        off = (off + bytes + 255) & ~(size_t)255;
        return o;
    };
    float*  deg    = (float*) (ws + carve((size_t)N_NODES * 4));
    float*  dinv   = (float*) (ws + carve((size_t)N_NODES * 4));
    __bf16* xbf    = (__bf16*)(ws + carve((size_t)N_NODES * IN_C * 2));
    __bf16* w1t    = (__bf16*)(ws + carve((size_t)IN_C * HID1 * 2));
    __bf16* w2t    = (__bf16*)(ws + carve((size_t)HID1 * HID2 * 2));
    __bf16* hbf    = (__bf16*)(ws + carve((size_t)N_NODES * HID1 * 2));
    float*  agg1   = (float*) (ws + carve((size_t)N_NODES * HID1 * 4));
    __bf16* h1bf   = (__bf16*)(ws + carve((size_t)N_NODES * HID1 * 2));
    __bf16* zprebf = (__bf16*)(ws + carve((size_t)N_NODES * HID2 * 2));
    float*  zf     = (float*) (ws + carve((size_t)N_NODES * HID2 * 4));
    (void)ws_size; (void)n_in; (void)in_sizes; (void)out_size;

    const int B = 256;
    auto gridFor = [](long long n, int b) { return (unsigned)((n + b - 1) / b); };

    // degrees + normalization
    k_deg_init <<<gridFor(N_NODES, B), B, 0, stream>>>(deg);
    k_deg_edges<<<gridFor(N_EDGES, B), B, 0, stream>>>(ei, deg);
    k_dinv     <<<gridFor(N_NODES, B), B, 0, stream>>>(deg, dinv);

    // bf16 conversions
    k_cvt_bf16      <<<gridFor((long long)N_NODES * IN_C, B), B, 0, stream>>>(x, xbf, N_NODES * IN_C);
    k_transpose_cvt <<<gridFor(IN_C * HID1, B), B, 0, stream>>>(W1, w1t, IN_C, HID1);
    k_transpose_cvt <<<gridFor(HID1 * HID2, B), B, 0, stream>>>(W2, w2t, HID1, HID2);

    // GEMM1: hbf[50000,256] = xbf @ W1  (tiles: 3125 x 16 = 50000 waves)
    {
        const int tilesM = N_NODES / 16, totalTiles = tilesM * (HID1 / 16);
        k_wmma_gemm<IN_C, HID1><<<gridFor((long long)totalTiles * 32, B), B, 0, stream>>>(
            xbf, w1t, hbf, tilesM, totalTiles);
    }

    // layer-1 aggregation (f32 atomics; working set resident in 192MB L2)
    k_agg1_init <<<gridFor((long long)N_NODES * HID1, B), B, 0, stream>>>(hbf, dinv, agg1);
    k_agg1_edges<<<gridFor((long long)N_EDGES * 32, B), B, 0, stream>>>(ei, dinv, hbf, agg1);
    k_post1     <<<gridFor((long long)N_NODES * HID1, B), B, 0, stream>>>(agg1, b1, h1bf);

    // GEMM2: zprebf[50000,32] = h1bf @ W2  (tiles: 3125 x 2 = 6250 waves)
    {
        const int tilesM = N_NODES / 16, totalTiles = tilesM * (HID2 / 16);
        k_wmma_gemm<HID1, HID2><<<gridFor((long long)totalTiles * 32, B), B, 0, stream>>>(
            h1bf, w2t, zprebf, tilesM, totalTiles);
    }

    // layer-2 aggregation
    k_agg2_init <<<gridFor((long long)N_NODES * HID2, B), B, 0, stream>>>(zprebf, dinv, b2, zf);
    k_agg2_edges<<<gridFor((long long)N_EDGES * 32, B), B, 0, stream>>>(ei, dinv, zprebf, zf);

    // decode
    k_decode<<<gridFor(N_LABEL, B), B, 0, stream>>>(eli, zf, out);
}
